// HybridMoE_14826227106476
// MI455X (gfx1250) — matrix operand order repaired and checked by reading, not compiled, not run on hardware
//
#include <hip/hip_runtime.h>
#include <hip/hip_bf16.h>

#define NUM_EXPERTS 16
#define TOPK 2
#define HIDDEN 2048
#define INTER 1408
#define TOKENS 4096
#define MAXPAIRS (TOKENS * TOPK)

// WMMA fragment types (probe-confirmed signatures for gfx1250 / clang-22)
typedef __attribute__((ext_vector_type(16))) __bf16          v16bf;
typedef __attribute__((ext_vector_type(8)))  float           v8f;
typedef __attribute__((ext_vector_type(16))) unsigned short  v16us;
typedef __attribute__((ext_vector_type(8)))  unsigned short  v8us;
typedef __attribute__((ext_vector_type(4)))  int             v4i;

static __device__ __forceinline__ unsigned short f2bf(float f) {
  unsigned u = __builtin_bit_cast(unsigned, f);
  u += 0x7fffu + ((u >> 16) & 1u);   // round-to-nearest-even
  return (unsigned short)(u >> 16);
}

static __device__ __forceinline__ v8f wmma_bf16(v16us a, v16us b, v8f c) {
  return __builtin_amdgcn_wmma_f32_16x16x32_bf16(
      /*neg_a=*/false, __builtin_bit_cast(v16bf, a),
      /*neg_b=*/false, __builtin_bit_cast(v16bf, b),
      /*c_mod=*/(short)0, c, /*reuse_a=*/false, /*reuse_b=*/false);
}

static __device__ __forceinline__ v16us ld_frag(const unsigned short* p0,
                                                const unsigned short* p1) {
  v8us a = *(const v8us*)p0;
  v8us b = *(const v8us*)p1;
  return __builtin_shufflevector(a, b, 0, 1, 2, 3, 4, 5, 6, 7,
                                 8, 9, 10, 11, 12, 13, 14, 15);
}

// ---- CDNA5 async LDS-direct copy path (ASYNCcnt) with sync fallback -------
// Signature (from compiler diagnostic): (v4i AS1* src, v4i AS3* dst, Ii, Ii)
#if defined(__has_builtin)
#if __has_builtin(__builtin_amdgcn_global_load_async_to_lds_b128)
#define USE_ASYNC_LDS 1
#endif
#endif

static __device__ __forceinline__ void copy16B_to_lds(const unsigned short* g,
                                                      unsigned short* l) {
#ifdef USE_ASYNC_LDS
  __builtin_amdgcn_global_load_async_to_lds_b128(
      (__attribute__((address_space(1))) v4i*)(g),
      (__attribute__((address_space(3))) v4i*)(l),
      /*imm_offset=*/0, /*cpol=*/0);
#else
  *(uint4*)l = *(const uint4*)g;
#endif
}

static __device__ __forceinline__ void async_wait_all() {
#ifdef USE_ASYNC_LDS
#if __has_builtin(__builtin_amdgcn_s_wait_asynccnt)
  __builtin_amdgcn_s_wait_asynccnt(0);
#else
  asm volatile("s_wait_asynccnt 0x0" ::: "memory");
#endif
#endif
}

// ---------------------------------------------------------------------------
// 1) Routing: top-2 of 16 logits, softmax over the pair, build per-expert
//    token lists. Entry id = 2*t + k  (fixed activation-row slot per pair).
// ---------------------------------------------------------------------------
__global__ void route_kernel(const float* __restrict__ logits,
                             int* __restrict__ counts,
                             int* __restrict__ lists,   // [E][TOKENS]
                             float* __restrict__ tw) {  // [MAXPAIRS]
  int t = blockIdx.x * blockDim.x + threadIdx.x;
  if (t >= TOKENS) return;
  const float* l = logits + t * NUM_EXPERTS;
  float v0 = -3.4e38f, v1 = -3.4e38f;
  int i0 = 0, i1 = 0;
#pragma unroll
  for (int i = 0; i < NUM_EXPERTS; ++i) {
    float v = l[i];
    if (v > v0) { v1 = v0; i1 = i0; v0 = v; i0 = i; }
    else if (v > v1) { v1 = v; i1 = i; }
  }
  float w0 = 1.f / (1.f + __expf(v1 - v0));   // softmax over {v0, v1}
  float w1 = 1.f - w0;
  tw[2 * t + 0] = w0;
  tw[2 * t + 1] = w1;
  int p0 = atomicAdd(&counts[i0], 1);
  lists[i0 * TOKENS + p0] = 2 * t + 0;
  int p1 = atomicAdd(&counts[i1], 1);
  lists[i1 * TOKENS + p1] = 2 * t + 1;
}

// ---------------------------------------------------------------------------
// 2) hidden fp32 -> bf16 staging (read activations once at full width)
// ---------------------------------------------------------------------------
__global__ void cvt_hidden(const float* __restrict__ x,
                           unsigned short* __restrict__ xb) {
  int i = (blockIdx.x * blockDim.x + threadIdx.x) * 4;
  float4 f = *(const float4*)(x + i);
  ushort4 h;
  h.x = f2bf(f.x); h.y = f2bf(f.y); h.z = f2bf(f.z); h.w = f2bf(f.w);
  *(ushort4*)(xb + i) = h;
}

// Tile geometry shared by both GEMM kernels
#define BM 16
#define BN 128
#define BK 64         // two 16x16x32 WMMA K-steps per LDS round-trip
#define PITCH 72      // LDS row pitch in elements (144B, 16B-aligned vectors)

// ---------------------------------------------------------------------------
// 3) GEMM1: for expert e, token tile m0..m0+15 (gathered), N-chunk of 128
//    gate & up columns.  a = silu(x@Wg) * (x@Wu), stored bf16 to act[].
//    grid = (INTER/BN, maxTiles, E), block = 256 (8 waves)
// ---------------------------------------------------------------------------
__global__ __launch_bounds__(256)
void ffn1_kernel(const unsigned short* __restrict__ hb,   // [T][H] bf16
                 const float* __restrict__ wg,            // [E][H][I]
                 const float* __restrict__ wu,            // [E][H][I]
                 const int* __restrict__ counts,
                 const int* __restrict__ lists,
                 unsigned short* __restrict__ act) {      // [MAXPAIRS][I] bf16
  const int e   = blockIdx.z;
  const int cnt = counts[e];
  const int m0  = blockIdx.y * BM;
  if (m0 >= cnt) return;
  const int n0  = blockIdx.x * BN;

  __shared__ __align__(16) unsigned short sA [BM * PITCH];
  __shared__ __align__(16) unsigned short sBg[BN * PITCH];
  __shared__ __align__(16) unsigned short sBu[BN * PITCH];
  __shared__ int sEnt[BM];

  const int tid  = threadIdx.x;
  const int lane = tid & 31;
  const int wave = tid >> 5;

  if (tid < BM) {
    int slot = m0 + tid;
    sEnt[tid] = (slot < cnt) ? lists[e * TOKENS + slot] : -1;
  }
  __syncthreads();

  // A staging: 128 threads, row = tid>>3, 8-elem (16B) chunk = tid&7
  const int arow = tid >> 3, achk = tid & 7;
  const int aent = (tid < 128) ? sEnt[arow] : -1;
  const unsigned short* asrc =
      (aent >= 0) ? (hb + (size_t)(aent >> 1) * HIDDEN + achk * 8) : nullptr;
  unsigned short* adst = &sA[arow * PITCH + achk * 8];
  if (tid < 128 && aent < 0) {           // padding rows: zero once, keep zero
    uint4 z = {0, 0, 0, 0};
    *(uint4*)adst = z;
  }

  // B staging: col-group (4 cols) = tid&31, base row = tid>>5 (+8*j, j<8)
  const int bcg = tid & 31;
  const int bk0 = tid >> 5;
  const float* wgbase = wg + (size_t)e * HIDDEN * INTER + n0 + bcg * 4;
  const float* wubase = wu + (size_t)e * HIDDEN * INTER + n0 + bcg * 4;

  // Fragment addressing (K contiguous per lane half, same order for A and B)
  const int mrow = lane & 15;               // A row
  const int kb_a = (lane >> 4) * 16;        // A: one 16-elem K chunk
  const int col  = wave * 16 + (lane & 15); // B/C column within tile
  const int kb_b = (lane >> 4) * 16;        // B: one 16-elem K chunk

  v8f cg = {};
  v8f cu = {};

  for (int k0 = 0; k0 < HIDDEN; k0 += BK) {
    if (tid < 128 && aent >= 0) {
      copy16B_to_lds(asrc + k0, adst);     // async LDS-direct on gfx1250
    }
#pragma unroll
    for (int j = 0; j < 8; ++j) {
      const int kk = bk0 + j * 8;
      const size_t roff = (size_t)(k0 + kk) * INTER;
      float4 g4 = *(const float4*)(wgbase + roff);
      float4 u4 = *(const float4*)(wubase + roff);
      sBg[(bcg * 4 + 0) * PITCH + kk] = f2bf(g4.x);
      sBg[(bcg * 4 + 1) * PITCH + kk] = f2bf(g4.y);
      sBg[(bcg * 4 + 2) * PITCH + kk] = f2bf(g4.z);
      sBg[(bcg * 4 + 3) * PITCH + kk] = f2bf(g4.w);
      sBu[(bcg * 4 + 0) * PITCH + kk] = f2bf(u4.x);
      sBu[(bcg * 4 + 1) * PITCH + kk] = f2bf(u4.y);
      sBu[(bcg * 4 + 2) * PITCH + kk] = f2bf(u4.z);
      sBu[(bcg * 4 + 3) * PITCH + kk] = f2bf(u4.w);
    }
    if (k0 + BK < HIDDEN) {   // L2 prefetch of next weight tile
      __builtin_prefetch(wgbase + (size_t)(k0 + BK + bk0) * INTER, 0, 1);
      __builtin_prefetch(wubase + (size_t)(k0 + BK + bk0) * INTER, 0, 1);
    }
    async_wait_all();
    __syncthreads();

#pragma unroll
    for (int s = 0; s < 2; ++s) {          // two 16x16x32 K-steps
      const unsigned short* ap = &sA[mrow * PITCH + s * 32 + kb_a];
      v16us af = ld_frag(ap, ap + 8);
      const unsigned short* gp = &sBg[col * PITCH + s * 32 + kb_b];
      v16us gf = ld_frag(gp, gp + 8);
      const unsigned short* up = &sBu[col * PITCH + s * 32 + kb_b];
      v16us uf = ld_frag(up, up + 8);
      cg = wmma_bf16(af, gf, cg);
      cu = wmma_bf16(af, uf, cu);
    }
    __syncthreads();
  }

  // Epilogue: silu(g)*u, bf16 store (C layout: VGPR r -> M = r + 8*(lane>=16),
  // N = lane&15)
  const int n_out = n0 + col;
#pragma unroll
  for (int r = 0; r < 8; ++r) {
    const int m = r + (lane >> 4) * 8;
    const int ent = sEnt[m];
    if (ent >= 0) {
      float g = cg[r], u = cu[r];
      float a = (g / (1.f + __expf(-g))) * u;
      act[(size_t)ent * INTER + n_out] = f2bf(a);
    }
  }
}

// ---------------------------------------------------------------------------
// 4) GEMM2: y = a @ Wd[e], weighted scatter-add to out via f32 atomics.
//    grid = (HIDDEN/BN, maxTiles, E), block = 256
// ---------------------------------------------------------------------------
__global__ __launch_bounds__(256)
void ffn2_kernel(const unsigned short* __restrict__ act,  // [MAXPAIRS][I] bf16
                 const float* __restrict__ wd,            // [E][I][H]
                 const int* __restrict__ counts,
                 const int* __restrict__ lists,
                 const float* __restrict__ tw,
                 float* __restrict__ out) {               // [T][H] f32 (zeroed)
  const int e   = blockIdx.z;
  const int cnt = counts[e];
  const int m0  = blockIdx.y * BM;
  if (m0 >= cnt) return;
  const int n0  = blockIdx.x * BN;

  __shared__ __align__(16) unsigned short sA[BM * PITCH];
  __shared__ __align__(16) unsigned short sB[BN * PITCH];
  __shared__ int sEnt[BM];

  const int tid  = threadIdx.x;
  const int lane = tid & 31;
  const int wave = tid >> 5;

  if (tid < BM) {
    int slot = m0 + tid;
    sEnt[tid] = (slot < cnt) ? lists[e * TOKENS + slot] : -1;
  }
  __syncthreads();

  const int arow = tid >> 3, achk = tid & 7;
  const int aent = (tid < 128) ? sEnt[arow] : -1;
  const unsigned short* asrc =
      (aent >= 0) ? (act + (size_t)aent * INTER + achk * 8) : nullptr;
  unsigned short* adst = &sA[arow * PITCH + achk * 8];
  if (tid < 128 && aent < 0) {
    uint4 z = {0, 0, 0, 0};
    *(uint4*)adst = z;
  }

  const int bcg = tid & 31;
  const int bk0 = tid >> 5;
  const float* wdbase = wd + (size_t)e * INTER * HIDDEN + n0 + bcg * 4;

  const int mrow = lane & 15;
  const int kb_a = (lane >> 4) * 16;
  const int col  = wave * 16 + (lane & 15);
  const int kb_b = (lane >> 4) * 16;

  v8f c = {};

  for (int k0 = 0; k0 < INTER; k0 += BK) {
    if (tid < 128 && aent >= 0) {
      copy16B_to_lds(asrc + k0, adst);
    }
#pragma unroll
    for (int j = 0; j < 8; ++j) {
      const int kk = bk0 + j * 8;
      float4 d4 = *(const float4*)(wdbase + (size_t)(k0 + kk) * HIDDEN);
      sB[(bcg * 4 + 0) * PITCH + kk] = f2bf(d4.x);
      sB[(bcg * 4 + 1) * PITCH + kk] = f2bf(d4.y);
      sB[(bcg * 4 + 2) * PITCH + kk] = f2bf(d4.z);
      sB[(bcg * 4 + 3) * PITCH + kk] = f2bf(d4.w);
    }
    if (k0 + BK < INTER) {
      __builtin_prefetch(wdbase + (size_t)(k0 + BK + bk0) * HIDDEN, 0, 1);
    }
    async_wait_all();
    __syncthreads();

#pragma unroll
    for (int s = 0; s < 2; ++s) {
      const unsigned short* ap = &sA[mrow * PITCH + s * 32 + kb_a];
      v16us af = ld_frag(ap, ap + 8);
      const unsigned short* bp = &sB[col * PITCH + s * 32 + kb_b];
      v16us bf = ld_frag(bp, bp + 8);
      c = wmma_bf16(af, bf, c);
    }
    __syncthreads();
  }

  const int n_out = n0 + col;
#pragma unroll
  for (int r = 0; r < 8; ++r) {
    const int m = r + (lane >> 4) * 8;
    const int ent = sEnt[m];
    if (ent >= 0) {
      const int t = ent >> 1;
      const float w = tw[ent];
      atomicAdd(&out[(size_t)t * HIDDEN + n_out], w * c[r]);
    }
  }
}

// ---------------------------------------------------------------------------
// Launch
// ---------------------------------------------------------------------------
extern "C" void kernel_launch(void* const* d_in, const int* in_sizes, int n_in,
                              void* d_out, int out_size, void* d_ws,
                              size_t ws_size, hipStream_t stream) {
  const float* hidden = (const float*)d_in[0];
  const float* logits = (const float*)d_in[1];
  const float* wg     = (const float*)d_in[2];
  const float* wu     = (const float*)d_in[3];
  const float* wd     = (const float*)d_in[4];
  float* out = (float*)d_out;

  char* ws = (char*)d_ws;
  size_t off = 0;
  int* counts = (int*)(ws + off);            off += 256;
  int* lists  = (int*)(ws + off);            off += (size_t)NUM_EXPERTS * TOKENS * sizeof(int);
  float* tw   = (float*)(ws + off);          off += (size_t)MAXPAIRS * sizeof(float);
  off = (off + 255) & ~(size_t)255;
  unsigned short* hb  = (unsigned short*)(ws + off);
  off += (size_t)TOKENS * HIDDEN * sizeof(unsigned short);
  unsigned short* act = (unsigned short*)(ws + off);

  (void)hipMemsetAsync(counts, 0, 256, stream);
  (void)hipMemsetAsync(d_out, 0, (size_t)out_size * sizeof(float), stream);

  cvt_hidden<<<(TOKENS * HIDDEN / 4) / 256, 256, 0, stream>>>(hidden, hb);
  route_kernel<<<TOKENS / 256, 256, 0, stream>>>(logits, counts, lists, tw);

  dim3 g1(INTER / BN, TOKENS / BM, NUM_EXPERTS);   // (11, 256, 16)
  ffn1_kernel<<<g1, 256, 0, stream>>>(hb, wg, wu, counts, lists, act);

  dim3 g2(HIDDEN / BN, TOKENS / BM, NUM_EXPERTS);  // (16, 256, 16)
  ffn2_kernel<<<g2, 256, 0, stream>>>(act, wd, counts, lists, tw, out);
}
